// SelfAttention_39505109188723
// MI455X (gfx1250) — compile-verified, matrix operand order checked by reading, not run on hardware
//
#include <hip/hip_runtime.h>
#include <hip/hip_bf16.h>

// MI455X (gfx1250) masked MHA:
//   k0: mask (N,1,S,S) int32 -> 1-bit pack        (kills 4.3 GB of mask re-reads)
//   kc: weights f32 -> f16 (once; keeps GEMM K-loops free of v_cvt chains)
//   k1: f16 QKV projection via v_wmma_f32_16x16x32_f16 (V stored transposed)
//   k2: flash attention: WMMA QK^T + online softmax + WMMA P*V (no score tensor)
//   k3: output projection GEMM (WMMA) + bias, f32 out
// E=1024 H=16 D=64 N=4 S=2048, softmax scale = 1/sqrt(E) = 1/32.

#define SLEN 2048
#define NH   16
#define HD   64
#define EMB  1024
#define NBAT 4

typedef __attribute__((ext_vector_type(16))) _Float16 v16h;
typedef __attribute__((ext_vector_type(8)))  _Float16 v8h;
typedef __attribute__((ext_vector_type(8)))  float    v8f;
typedef __attribute__((ext_vector_type(4)))  float    v4f;

// One WMMA f16 A/B fragment for a lane = halves {p[0..7], p[16..23]}.
// Caller pre-offsets p by kko (0 for lanes 0-15, 8 for lanes 16-31) and k0.
static __device__ __forceinline__ v16h ldfrag_h(const _Float16* p) {
  v8h lo = *(const v8h*)(p);
  v8h hi = *(const v8h*)(p + 16);
  return __builtin_shufflevector(lo, hi, 0,1,2,3,4,5,6,7,8,9,10,11,12,13,14,15);
}

// Same fragment gathered from f32 memory with on-the-fly cvt to f16
// (used only for activations, which are touched exactly once).
static __device__ __forceinline__ v16h ldfrag_f(const float* p) {
  v4f a = *(const v4f*)(p);
  v4f b = *(const v4f*)(p + 4);
  v4f c = *(const v4f*)(p + 16);
  v4f d = *(const v4f*)(p + 20);
  v16h r;
#pragma unroll
  for (int i = 0; i < 4; ++i) {
    r[i]      = (_Float16)a[i];
    r[4 + i]  = (_Float16)b[i];
    r[8 + i]  = (_Float16)c[i];
    r[12 + i] = (_Float16)d[i];
  }
  return r;
}

#define WMMA_F16(a, b, c) \
  __builtin_amdgcn_wmma_f32_16x16x32_f16(false, (a), false, (b), (short)0, (c), false, false)

// ---------------------------------------------------------------------------
// Kernel c: one-shot f32 -> f16 weight conversion.
// ---------------------------------------------------------------------------
__global__ void __launch_bounds__(256) cvt_f16_kernel(
    const float* __restrict__ src, _Float16* __restrict__ dst, int n) {
  const int i = blockIdx.x * 256 + threadIdx.x;
  if (i < n) dst[i] = (_Float16)src[i];
}

// ---------------------------------------------------------------------------
// Kernel 0: pack int32 mask into bitmask words: bit b of word w of row (n,q)
// = mask[n][0][q][w*32+b] != 0.  One wave -> one word via ballot.
// ---------------------------------------------------------------------------
__global__ void __launch_bounds__(256) pack_mask_kernel(
    const int* __restrict__ mask, unsigned* __restrict__ mpk) {
  const int wave = threadIdx.x >> 5, lane = threadIdx.x & 31;
  const long long gid = (long long)blockIdx.x * 8 + wave;
  const int pred = mask[gid * 32 + lane] != 0;
  unsigned long long bal = __ballot(pred);
  if (lane == 0) mpk[gid] = (unsigned)bal;
}

// ---------------------------------------------------------------------------
// Kernel 1: per-head projections.  X viewed as (N*S*H, 64) contiguous rows,
// out[r][j] = sum_k X[r][k] * W[j][k].  One wave computes a 16x64 tile with
// 8 chained WMMAs.  Q,K stored row-major f16 (N,S,H,D); V stored TRANSPOSED
// f16 (N,H,D,S) so flash P*V B-fragments become contiguous 128-bit loads.
// ---------------------------------------------------------------------------
__global__ void __launch_bounds__(256) qkv_proj_kernel(
    const float* __restrict__ xq, const float* __restrict__ xk,
    const float* __restrict__ xv,
    const _Float16* __restrict__ Wq, const _Float16* __restrict__ Wk,
    const _Float16* __restrict__ Wv,
    _Float16* __restrict__ qh, _Float16* __restrict__ kh,
    _Float16* __restrict__ vT) {
  const int wave = threadIdx.x >> 5, lane = threadIdx.x & 31;
  const int ncol = lane & 15, hi = lane >> 4, kko = hi * 8;
  const int mtile = blockIdx.x * 8 + wave;

  const float* x;
  const _Float16* W;
  if (blockIdx.y == 0)      { x = xq; W = Wq; }
  else if (blockIdx.y == 1) { x = xk; W = Wk; }
  else                      { x = xv; W = Wv; }

  const float* xrow = x + (size_t)(mtile * 16 + ncol) * HD;
  const v16h a0 = ldfrag_f(xrow + kko);
  const v16h a1 = ldfrag_f(xrow + 32 + kko);

  v8f acc[4];
#pragma unroll
  for (int ns = 0; ns < 4; ++ns) {
    const _Float16* wrow = W + (size_t)(ns * 16 + ncol) * HD;  // col of W^T = row of W
    v8f c = {};
    c = WMMA_F16(a0, ldfrag_h(wrow + kko), c);
    c = WMMA_F16(a1, ldfrag_h(wrow + 32 + kko), c);
    acc[ns] = c;
  }

  if (blockIdx.y < 2) {
    _Float16* out = (blockIdx.y == 0) ? qh : kh;
#pragma unroll
    for (int ns = 0; ns < 4; ++ns)
#pragma unroll
      for (int r = 0; r < 8; ++r) {
        const int m = r + 8 * hi;
        out[(size_t)(mtile * 16 + m) * HD + ns * 16 + ncol] = (_Float16)acc[ns][r];
      }
  } else {
#pragma unroll
    for (int ns = 0; ns < 4; ++ns)
#pragma unroll
      for (int r = 0; r < 8; ++r) {
        const int m = r + 8 * hi;
        const int R = mtile * 16 + m;        // R = (n*S + s)*H + h
        const int h = R & 15;
        const int s = (R >> 4) & (SLEN - 1);
        const int nb = R >> 15;
        const int d = ns * 16 + ncol;
        vT[(((size_t)nb * NH + h) * HD + d) * SLEN + s] = (_Float16)acc[ns][r];
      }
  }
}

// ---------------------------------------------------------------------------
// Kernel 2: flash attention.  One wave owns a 16-row Q tile for one (n,h).
// Per 32-key step: 2 score tiles (2 WMMAs each), packed-mask apply, online
// softmax (shfl_xor reductions within 16-lane groups), P transposed through
// a 1 KB per-wave LDS buffer, then 4 WMMAs of P(16x32) x V(32x64).
// Next iteration's K/V streams are prefetched (global_prefetch_b8).
// ---------------------------------------------------------------------------
__global__ void __launch_bounds__(256) flash_attn_kernel(
    const _Float16* __restrict__ qh, const _Float16* __restrict__ kh,
    const _Float16* __restrict__ vT, const unsigned* __restrict__ mpk,
    _Float16* __restrict__ aout) {
  __shared__ __align__(16) _Float16 Pbuf[8][16][32];

  const int wave = threadIdx.x >> 5, lane = threadIdx.x & 31;
  const int ncol = lane & 15, hi = lane >> 4, kko = hi * 8;
  const int h = blockIdx.y, n = blockIdx.z;
  const int qbase = (blockIdx.x * 8 + wave) * 16;

  const _Float16* qrow = qh + (((size_t)n * SLEN + qbase + ncol) * NH + h) * HD;
  const v16h aQ0 = ldfrag_h(qrow + kko);
  const v16h aQ1 = ldfrag_h(qrow + 32 + kko);

  float mrow[8], lrow[8];
  v8f O[4];
#pragma unroll
  for (int r = 0; r < 8; ++r) { mrow[r] = -3.0e38f; lrow[r] = 0.0f; }
#pragma unroll
  for (int ns = 0; ns < 4; ++ns) { v8f z = {}; O[ns] = z; }

  const unsigned* mrowp = mpk + ((size_t)n * SLEN + qbase) * (SLEN / 32);
  const float scale = 0.03125f;        // 1/sqrt(1024)
  const float L2E = 1.44269504f;

#pragma unroll 1
  for (int j = 0; j < SLEN; j += 32) {
    // --- scores for key sub-tiles [j..j+15] and [j+16..j+31]
    const _Float16* krow0 = kh + (((size_t)n * SLEN + j + ncol) * NH + h) * HD;
    const _Float16* krow1 = krow0 + (size_t)16 * NH * HD;
    // prefetch next key step's rows (same lane -> same row, +32 keys ahead)
    if (j + 32 < SLEN) {
      __builtin_prefetch(krow0 + (size_t)32 * NH * HD, 0, 1);
      __builtin_prefetch(krow1 + (size_t)32 * NH * HD, 0, 1);
    }
    v8f c0 = {}, c1 = {};
    c0 = WMMA_F16(aQ0, ldfrag_h(krow0 + kko), c0);
    c0 = WMMA_F16(aQ1, ldfrag_h(krow0 + 32 + kko), c0);
    c1 = WMMA_F16(aQ0, ldfrag_h(krow1 + kko), c1);
    c1 = WMMA_F16(aQ1, ldfrag_h(krow1 + 32 + kko), c1);

    // --- packed mask + online softmax (per q-row r, reduce across 16 lanes)
    const int wword = j >> 5;
#pragma unroll
    for (int r = 0; r < 8; ++r) {
      const unsigned w = mrowp[(size_t)(r + 8 * hi) * (SLEN / 32) + wword];
      float s0 = ((w >> ncol) & 1u)        ? c0[r] * scale : -3.0e18f;
      float s1 = ((w >> (ncol + 16)) & 1u) ? c1[r] * scale : -3.0e18f;

      float mx = fmaxf(s0, s1);
#pragma unroll
      for (int off = 1; off < 16; off <<= 1)
        mx = fmaxf(mx, __shfl_xor(mx, off, 32));
      const float mnew = fmaxf(mrow[r], mx);
      const float f = exp2f((mrow[r] - mnew) * L2E);
      const float p0 = exp2f((s0 - mnew) * L2E);
      const float p1 = exp2f((s1 - mnew) * L2E);
      float psum = p0 + p1;
#pragma unroll
      for (int off = 1; off < 16; off <<= 1)
        psum += __shfl_xor(psum, off, 32);

      lrow[r] = lrow[r] * f + psum;
      mrow[r] = mnew;
#pragma unroll
      for (int ns = 0; ns < 4; ++ns) O[ns][r] *= f;   // rescale accumulator

      Pbuf[wave][r + 8 * hi][ncol]      = (_Float16)p0;  // C-layout -> row-major
      Pbuf[wave][r + 8 * hi][ncol + 16] = (_Float16)p1;
    }

    // --- O += P(16x32) x V(32x64); P A-frag from LDS, V B-frags from vT
    const v16h aP = ldfrag_h(&Pbuf[wave][ncol][kko]);
#pragma unroll
    for (int ns = 0; ns < 4; ++ns) {
      const _Float16* vrow =
          vT + (((size_t)n * NH + h) * HD + ns * 16 + ncol) * SLEN + j;
      if (j + 32 < SLEN) __builtin_prefetch(vrow + 32, 0, 1);
      O[ns] = WMMA_F16(aP, ldfrag_h(vrow + kko), O[ns]);
    }
  }

  // --- epilogue: normalize and store f16 attention output (N,S,E)
  float inv[8];
#pragma unroll
  for (int r = 0; r < 8; ++r) inv[r] = 1.0f / lrow[r];
#pragma unroll
  for (int ns = 0; ns < 4; ++ns)
#pragma unroll
    for (int r = 0; r < 8; ++r) {
      const int q = qbase + r + 8 * hi;
      aout[((size_t)n * SLEN + q) * EMB + h * HD + ns * 16 + ncol] =
          (_Float16)(O[ns][r] * inv[r]);
    }
}

// ---------------------------------------------------------------------------
// Kernel 3: output projection  Y = A(8192x1024) * WoH^T + bo, f32 result.
// One wave -> 16x64 tile, K-loop 32 steps x 4 WMMAs; WoH is pre-converted f16
// so the hot loop is pure b128 loads + wmma (no cvt chains).
// ---------------------------------------------------------------------------
__global__ void __launch_bounds__(256) out_proj_kernel(
    const _Float16* __restrict__ aat, const _Float16* __restrict__ WoH,
    const float* __restrict__ bo, float* __restrict__ out) {
  const int wave = threadIdx.x >> 5, lane = threadIdx.x & 31;
  const int ncol = lane & 15, hi = lane >> 4, kko = hi * 8;
  const int mtile = blockIdx.x * 8 + wave;
  const int n0 = blockIdx.y * 64;

  const _Float16* arow = aat + (size_t)(mtile * 16 + ncol) * EMB;
  const _Float16* wrow[4];
#pragma unroll
  for (int ns = 0; ns < 4; ++ns)
    wrow[ns] = WoH + (size_t)(n0 + ns * 16 + ncol) * EMB;

  v8f acc[4];
#pragma unroll
  for (int ns = 0; ns < 4; ++ns) { v8f z = {}; acc[ns] = z; }

#pragma unroll 1
  for (int k0 = 0; k0 < EMB; k0 += 32) {
    const v16h a = ldfrag_h(arow + k0 + kko);
#pragma unroll
    for (int ns = 0; ns < 4; ++ns)
      acc[ns] = WMMA_F16(a, ldfrag_h(wrow[ns] + k0 + kko), acc[ns]);
  }

#pragma unroll
  for (int ns = 0; ns < 4; ++ns) {
    const float bias = bo[n0 + ns * 16 + ncol];
#pragma unroll
    for (int r = 0; r < 8; ++r) {
      const int m = r + 8 * hi;
      out[(size_t)(mtile * 16 + m) * EMB + n0 + ns * 16 + ncol] =
          acc[ns][r] + bias;
    }
  }
}

// ---------------------------------------------------------------------------
extern "C" void kernel_launch(void* const* d_in, const int* in_sizes, int n_in,
                              void* d_out, int out_size, void* d_ws,
                              size_t ws_size, hipStream_t stream) {
  const float* values  = (const float*)d_in[0];
  const float* keysx   = (const float*)d_in[1];
  const float* queries = (const float*)d_in[2];
  const int*   mask    = (const int*)d_in[3];
  const float* Wv = (const float*)d_in[4];
  const float* Wk = (const float*)d_in[5];
  const float* Wq = (const float*)d_in[6];
  const float* Wo = (const float*)d_in[7];
  const float* bo = (const float*)d_in[8];
  float* out = (float*)d_out;

  char* ws = (char*)d_ws;
  const size_t halfmat = (size_t)NBAT * SLEN * EMB * sizeof(_Float16);  // 16 MiB
  _Float16* qh  = (_Float16*)(ws);                  // (N,S,H,D) f16
  _Float16* kh  = (_Float16*)(ws + halfmat);        // (N,S,H,D) f16
  _Float16* vT  = (_Float16*)(ws + 2 * halfmat);    // (N,H,D,S) f16 (transposed)
  _Float16* aat = (_Float16*)(ws + 3 * halfmat);    // (N,S,E)   f16
  unsigned* mpk = (unsigned*)(ws + 4 * halfmat);    // (N,S,S/32) bitmask, 2 MiB
  char* wbase = ws + 4 * halfmat + (size_t)NBAT * SLEN * (SLEN / 32) * 4;
  _Float16* WoH = (_Float16*)(wbase);                           // 2 MiB
  _Float16* WqH = (_Float16*)(wbase + (size_t)EMB * EMB * 2);
  _Float16* WkH = WqH + HD * HD;
  _Float16* WvH = WkH + HD * HD;

  // c) one-shot weight conversions f32 -> f16
  cvt_f16_kernel<<<dim3((EMB * EMB + 255) / 256), dim3(256), 0, stream>>>(
      Wo, WoH, EMB * EMB);
  cvt_f16_kernel<<<dim3((HD * HD + 255) / 256), dim3(256), 0, stream>>>(
      Wq, WqH, HD * HD);
  cvt_f16_kernel<<<dim3((HD * HD + 255) / 256), dim3(256), 0, stream>>>(
      Wk, WkH, HD * HD);
  cvt_f16_kernel<<<dim3((HD * HD + 255) / 256), dim3(256), 0, stream>>>(
      Wv, WvH, HD * HD);
  // 0) mask -> bitmask (one wave per 32-bit word)
  pack_mask_kernel<<<dim3((NBAT * SLEN * SLEN / 32) / 8), dim3(256), 0, stream>>>(
      mask, mpk);
  // 1) QKV projections (grid.y selects q/k/v)
  qkv_proj_kernel<<<dim3((NBAT * SLEN * NH / 16) / 8, 3), dim3(256), 0, stream>>>(
      queries, keysx, values, WqH, WkH, WvH, qh, kh, vT);
  // 2) flash attention: 8 q-tiles per block, grid over (q-tiles, H, N)
  flash_attn_kernel<<<dim3(SLEN / 128, NH, NBAT), dim3(256), 0, stream>>>(
      qh, kh, vT, mpk, aat);
  // 3) output projection + bias
  out_proj_kernel<<<dim3((NBAT * SLEN / 16) / 8, EMB / 64), dim3(256), 0, stream>>>(
      aat, WoH, bo, out);
}
